// GNNBlock_69690139345480
// MI455X (gfx1250) — matrix operand order repaired and checked by reading, not compiled, not run on hardware
//
#include <hip/hip_runtime.h>

#define HDIM 512
#define BDIM 4
#define NDIM 10242
#define KNN  8
#define TN   8      // nodes per block
#define TE   64     // edges per block (TN * KNN)
#define NTILES 32   // HDIM / 16 column tiles

typedef __attribute__((ext_vector_type(16))) __bf16   bf16x16;
typedef __attribute__((ext_vector_type(8)))  float    f32x8;
typedef __attribute__((ext_vector_type(4)))  float    f32x4;
typedef __attribute__((ext_vector_type(4)))  unsigned u32x4;
typedef __attribute__((ext_vector_type(2)))  unsigned u32x2;

union Frag {
  u32x4   q[2];
  bf16x16 v;
};

__device__ __forceinline__ unsigned short f2bf(float f) {
  unsigned u = __float_as_uint(f);
  u += 0x7FFFu + ((u >> 16) & 1u);   // round-to-nearest-even
  return (unsigned short)(u >> 16);
}

// 16B async copy global -> LDS (GLOBAL_LOAD_ASYNC_TO_LDS_B128, GVS mode).
// lds_off: byte offset in LDS (generic-pointer low 32 bits); gofs: byte offset
// from base (signed 32-bit range is sufficient: ws regions < 2 GiB).
__device__ __forceinline__ void async_cp16(unsigned lds_off, unsigned gofs,
                                           const void* base) {
  asm volatile("global_load_async_to_lds_b128 %0, %1, %2"
               :
               : "v"(lds_off), "v"(gofs), "s"(base)
               : "memory");
}
__device__ __forceinline__ void async_wait0() {
  asm volatile("s_wait_asynccnt 0x0" ::: "memory");
}

// ---------------- prep: x -> bf16 ----------------
__global__ void cvt_x_kernel(const float* __restrict__ x,
                             unsigned short* __restrict__ xb, int n4) {
  int stride = gridDim.x * blockDim.x;
  for (int i = blockIdx.x * blockDim.x + threadIdx.x; i < n4; i += stride) {
    f32x4 f = ((const f32x4*)x)[i];
    u32x2 o;
    o.x = (unsigned)f2bf(f.x) | ((unsigned)f2bf(f.y) << 16);
    o.y = (unsigned)f2bf(f.z) | ((unsigned)f2bf(f.w) << 16);
    ((u32x2*)xb)[i] = o;
  }
}

// ------- prep: weights -> bf16 fragment-major (CDNA5 B-matrix layout) -------
__global__ void repack_w_kernel(const float* __restrict__ W,
                                unsigned short* __restrict__ Wf, int Kdim) {
  int total = Kdim * HDIM;
  int stride = gridDim.x * blockDim.x;
  for (int i = blockIdx.x * blockDim.x + threadIdx.x; i < total; i += stride) {
    int k = i >> 9;            // row (K index), HDIM = 512 cols
    int n = i & 511;           // col (N index)
    int kt = k >> 5, kr = k & 31;
    int hi = (kr >> 3) & 1;    // lane half
    int j  = ((kr & 7) | ((kr >> 4) << 3)) >> 1;   // VGPR index
    int h  = kr & 1;           // low/high half of VGPR
    int nt = n >> 4;
    int lane = (n & 15) | (hi << 4);
    int off = (((kt * NTILES + nt) * 32 + lane) << 4) + (j << 1) + h;
    Wf[off] = f2bf(W[i]);
  }
}

// ---------------- fused GNN block ----------------
__global__ __launch_bounds__(256) void gnn_fused_kernel(
    const float* __restrict__ x,
    const float* __restrict__ lat_bias,
    const long long* __restrict__ knn,
    const float* __restrict__ mb1,
    const float* __restrict__ mb2,
    const float* __restrict__ ub1,
    const float* __restrict__ ub2,
    const float* __restrict__ ln_g,
    const float* __restrict__ ln_b,
    const unsigned short* __restrict__ xb,
    const unsigned short* __restrict__ mW1f,
    const unsigned short* __restrict__ mW2f,
    const unsigned short* __restrict__ uW1f,
    const unsigned short* __restrict__ uW2f,
    float* __restrict__ out) {
  __shared__ __align__(16) unsigned char smem[147712];
  unsigned short* ctr  = (unsigned short*)(smem);            // [8][512] bf16
  unsigned short* nbr  = (unsigned short*)(smem + 8192);     // [64][512] bf16
  unsigned short* hid  = (unsigned short*)(smem + 73728);    // [64][512] bf16 (reused)
  unsigned short* msgs = (unsigned short*)(smem + 139264);   // [8][512] bf16
  int*   idxs = (int*)(smem + 147456);                       // [64]
  float* ybuf = (float*)(smem + 8192);                       // [8][512] f32, aliases nbr

  const int tiles = (NDIM + TN - 1) / TN;
  const int bidx = blockIdx.x / tiles;
  const int t = blockIdx.x - bidx * tiles;
  const int n0 = t * TN;
  const int tid = threadIdx.x;
  const int lane = tid & 31;
  const int wave = tid >> 5;
  const int ln15 = lane & 15;
  const int lhi  = lane >> 4;
  const size_t xrowbase = (size_t)bidx * NDIM;
  const f32x8 zero8 = {0.f, 0.f, 0.f, 0.f, 0.f, 0.f, 0.f, 0.f};

  // ---- stage knn indices ----
  if (tid < TE) {
    int i = tid >> 3, k = tid & 7;
    int node = n0 + i; if (node >= NDIM) node = NDIM - 1;
    idxs[tid] = (int)knn[(long long)node * KNN + k];
  }
  __syncthreads();

  // ---- stage centers + gathered neighbors via async global->LDS b128 ----
  const unsigned xb_row0 = (unsigned)(xrowbase * HDIM) * 2u;  // byte offset of batch
  const unsigned ctr_base = (unsigned)(size_t)ctr;            // LDS byte offsets
  const unsigned nbr_base = (unsigned)(size_t)nbr;
#pragma unroll
  for (int it = 0; it < 2; ++it) {                 // 8 rows * 64 chunks
    int c = tid + it * 256;
    int row = c >> 6, ch = c & 63;
    int node = n0 + row; if (node >= NDIM) node = NDIM - 1;
    unsigned gofs = xb_row0 + ((unsigned)node * HDIM + (unsigned)ch * 8u) * 2u;
    async_cp16(ctr_base + (unsigned)(row * HDIM + ch * 8) * 2u, gofs, xb);
  }
#pragma unroll 4
  for (int it = 0; it < 16; ++it) {                // 64 rows * 64 chunks
    int c = tid + it * 256;
    int e = c >> 6, ch = c & 63;
    int node = idxs[e];
    unsigned gofs = xb_row0 + ((unsigned)node * HDIM + (unsigned)ch * 8u) * 2u;
    async_cp16(nbr_base + (unsigned)(e * HDIM + ch * 8) * 2u, gofs, xb);
  }
  async_wait0();
  __syncthreads();

  // ---- phase B: message MLP layer 1, hidden[64][512] = silu([c|n]@mW1 + b1) ----
  for (int nti = 0; nti < 4; ++nti) {
    const int nt = wave + nti * 8;
    f32x8 acc[4] = {zero8, zero8, zero8, zero8};
    for (int kt = 0; kt < 16; ++kt) {              // center half (rows repeat 8x)
      Frag bf;
      const unsigned short* wp = mW1f + (((kt * NTILES + nt) * 32 + lane) << 4);
      bf.q[0] = *(const u32x4*)(wp);
      bf.q[1] = *(const u32x4*)(wp + 8);
#pragma unroll
      for (int mt = 0; mt < 4; ++mt) {
        Frag af;
        const unsigned short* ap = ctr + ((mt * 16 + ln15) >> 3) * HDIM + kt * 32 + lhi * 8;
        af.q[0] = *(const u32x4*)(ap);
        af.q[1] = *(const u32x4*)(ap + 16);
        acc[mt] = __builtin_amdgcn_wmma_f32_16x16x32_bf16(
            false, af.v, false, bf.v, (short)0, acc[mt], false, false);
      }
    }
    for (int kt = 16; kt < 32; ++kt) {             // neighbor half
      Frag bf;
      const unsigned short* wp = mW1f + (((kt * NTILES + nt) * 32 + lane) << 4);
      bf.q[0] = *(const u32x4*)(wp);
      bf.q[1] = *(const u32x4*)(wp + 8);
#pragma unroll
      for (int mt = 0; mt < 4; ++mt) {
        Frag af;
        const unsigned short* ap = nbr + (mt * 16 + ln15) * HDIM + (kt - 16) * 32 + lhi * 8;
        af.q[0] = *(const u32x4*)(ap);
        af.q[1] = *(const u32x4*)(ap + 16);
        acc[mt] = __builtin_amdgcn_wmma_f32_16x16x32_bf16(
            false, af.v, false, bf.v, (short)0, acc[mt], false, false);
      }
    }
    const int col = nt * 16 + ln15;
    const float b1 = mb1[col];
#pragma unroll
    for (int mt = 0; mt < 4; ++mt) {
#pragma unroll
      for (int r = 0; r < 8; ++r) {
        float v = acc[mt][r] + b1;
        v = v / (1.0f + __expf(-v));         // silu
        int m = mt * 16 + r + lhi * 8;
        hid[m * HDIM + col] = f2bf(v);
      }
    }
  }
  __syncthreads();

  // ---- phase C: message MLP layer 2 + mean over K (per-lane register reduce) ----
  for (int nti = 0; nti < 4; ++nti) {
    const int nt = wave + nti * 8;
    const int col = nt * 16 + ln15;
    const float b2 = mb2[col];
    for (int mt = 0; mt < 4; ++mt) {
      f32x8 acc = zero8;
      for (int kt = 0; kt < 16; ++kt) {
        Frag bf, af;
        const unsigned short* wp = mW2f + (((kt * NTILES + nt) * 32 + lane) << 4);
        bf.q[0] = *(const u32x4*)(wp);
        bf.q[1] = *(const u32x4*)(wp + 8);
        const unsigned short* ap = hid + (mt * 16 + ln15) * HDIM + kt * 32 + lhi * 8;
        af.q[0] = *(const u32x4*)(ap);
        af.q[1] = *(const u32x4*)(ap + 16);
        acc = __builtin_amdgcn_wmma_f32_16x16x32_bf16(
            false, af.v, false, bf.v, (short)0, acc, false, false);
      }
      // lane's 8 C rows are exactly the 8 edges of one node -> mean is free
      float s = 0.f;
#pragma unroll
      for (int r = 0; r < 8; ++r) s += acc[r];
      float mv = s * 0.125f + b2;
      int node = mt * 2 + lhi;
      msgs[node * HDIM + col] = f2bf(mv);
    }
  }
  __syncthreads();

  // ---- phase D: update MLP layer 1, hiddenU[16][512] (rows 8..15 dup) ----
  for (int nti = 0; nti < 4; ++nti) {
    const int nt = wave + nti * 8;
    f32x8 acc = zero8;
    const int arow = ln15 & 7;
    for (int kt = 0; kt < 16; ++kt) {              // x half
      Frag bf, af;
      const unsigned short* wp = uW1f + (((kt * NTILES + nt) * 32 + lane) << 4);
      bf.q[0] = *(const u32x4*)(wp);
      bf.q[1] = *(const u32x4*)(wp + 8);
      const unsigned short* ap = ctr + arow * HDIM + kt * 32 + lhi * 8;
      af.q[0] = *(const u32x4*)(ap);
      af.q[1] = *(const u32x4*)(ap + 16);
      acc = __builtin_amdgcn_wmma_f32_16x16x32_bf16(
          false, af.v, false, bf.v, (short)0, acc, false, false);
    }
    for (int kt = 16; kt < 32; ++kt) {             // messages half
      Frag bf, af;
      const unsigned short* wp = uW1f + (((kt * NTILES + nt) * 32 + lane) << 4);
      bf.q[0] = *(const u32x4*)(wp);
      bf.q[1] = *(const u32x4*)(wp + 8);
      const unsigned short* ap = msgs + arow * HDIM + (kt - 16) * 32 + lhi * 8;
      af.q[0] = *(const u32x4*)(ap);
      af.q[1] = *(const u32x4*)(ap + 16);
      acc = __builtin_amdgcn_wmma_f32_16x16x32_bf16(
          false, af.v, false, bf.v, (short)0, acc, false, false);
    }
    const int col = nt * 16 + ln15;
    const float b1 = ub1[col];
#pragma unroll
    for (int r = 0; r < 8; ++r) {
      float v = acc[r] + b1;
      v = v / (1.0f + __expf(-v));
      int m = r + lhi * 8;
      hid[m * HDIM + col] = f2bf(v);
    }
  }
  __syncthreads();

  // ---- phase E: update MLP layer 2 + residual + lat_bias -> ybuf f32 ----
  for (int nti = 0; nti < 4; ++nti) {
    const int nt = wave + nti * 8;
    f32x8 acc = zero8;
    for (int kt = 0; kt < 16; ++kt) {
      Frag bf, af;
      const unsigned short* wp = uW2f + (((kt * NTILES + nt) * 32 + lane) << 4);
      bf.q[0] = *(const u32x4*)(wp);
      bf.q[1] = *(const u32x4*)(wp + 8);
      const unsigned short* ap = hid + ln15 * HDIM + kt * 32 + lhi * 8;
      af.q[0] = *(const u32x4*)(ap);
      af.q[1] = *(const u32x4*)(ap + 16);
      acc = __builtin_amdgcn_wmma_f32_16x16x32_bf16(
          false, af.v, false, bf.v, (short)0, acc, false, false);
    }
    const int col = nt * 16 + ln15;
    if (lhi == 0) {                            // rows 0..7 are the real nodes
      const float b2 = ub2[col];
#pragma unroll
      for (int r = 0; r < 8; ++r) {
        int node = n0 + r; if (node >= NDIM) node = NDIM - 1;
        float xv = x[(xrowbase + node) * HDIM + col];
        float lb = lat_bias[(size_t)node * HDIM + col];
        ybuf[r * HDIM + col] = xv + acc[r] + b2 + lb;
      }
    }
  }
  __syncthreads();

  // ---- LayerNorm: one wave per node row ----
  {
    const int row = wave;
    float vals[16];
    float s = 0.f, ss = 0.f;
#pragma unroll
    for (int i = 0; i < 16; ++i) {
      float y = ybuf[row * HDIM + lane + i * 32];
      vals[i] = y; s += y; ss += y * y;
    }
#pragma unroll
    for (int off = 16; off > 0; off >>= 1) {
      s  += __shfl_xor(s,  off, 32);
      ss += __shfl_xor(ss, off, 32);
    }
    const float mu = s * (1.0f / HDIM);
    float var = ss * (1.0f / HDIM) - mu * mu;
    var = var < 0.f ? 0.f : var;
    const float rs = rsqrtf(var + 1e-5f);
    int node = n0 + row; if (node >= NDIM) node = NDIM - 1;
    float* op = out + (xrowbase + node) * HDIM;
#pragma unroll
    for (int i = 0; i < 16; ++i) {
      int col = lane + i * 32;
      op[col] = (vals[i] - mu) * rs * ln_g[col] + ln_b[col];
    }
  }
}

extern "C" void kernel_launch(void* const* d_in, const int* in_sizes, int n_in,
                              void* d_out, int out_size, void* d_ws, size_t ws_size,
                              hipStream_t stream) {
  (void)in_sizes; (void)n_in; (void)out_size; (void)ws_size;
  const float* x   = (const float*)d_in[0];
  const float* lat = (const float*)d_in[1];
  const long long* knn = (const long long*)d_in[2];
  const float* mW1 = (const float*)d_in[3];
  const float* mb1 = (const float*)d_in[4];
  const float* mW2 = (const float*)d_in[5];
  const float* mb2 = (const float*)d_in[6];
  const float* uW1 = (const float*)d_in[7];
  const float* ub1 = (const float*)d_in[8];
  const float* uW2 = (const float*)d_in[9];
  const float* ub2 = (const float*)d_in[10];
  const float* lng = (const float*)d_in[11];
  const float* lnb = (const float*)d_in[12];

  unsigned char* ws = (unsigned char*)d_ws;
  const size_t xb_bytes = (size_t)BDIM * NDIM * HDIM * 2;   // ~40 MB
  unsigned short* xb   = (unsigned short*)(ws);
  unsigned short* mW1f = (unsigned short*)(ws + xb_bytes);
  unsigned short* mW2f = (unsigned short*)(ws + xb_bytes + 1048576);
  unsigned short* uW1f = (unsigned short*)(ws + xb_bytes + 1048576 + 524288);
  unsigned short* uW2f = (unsigned short*)(ws + xb_bytes + 1048576 + 524288 + 1048576);

  const int n4 = BDIM * NDIM * HDIM / 4;
  cvt_x_kernel<<<2048, 256, 0, stream>>>(x, xb, n4);
  repack_w_kernel<<<1024, 256, 0, stream>>>(mW1, mW1f, 2 * HDIM);
  repack_w_kernel<<<512,  256, 0, stream>>>(mW2, mW2f, HDIM);
  repack_w_kernel<<<1024, 256, 0, stream>>>(uW1, uW1f, 2 * HDIM);
  repack_w_kernel<<<512,  256, 0, stream>>>(uW2, uW2f, HDIM);

  const int tiles = (NDIM + TN - 1) / TN;
  gnn_fused_kernel<<<BDIM * tiles, 256, 0, stream>>>(
      x, lat, knn, mb1, mb2, ub1, ub2, lng, lnb,
      xb, mW1f, mW2f, uW1f, uW2f, (float*)d_out);
}
